// CQAttention_12352325943422
// MI455X (gfx1250) — matrix operand order repaired and checked by reading, not compiled, not run on hardware
//
#include <hip/hip_runtime.h>
#include <hip/hip_bf16.h>
#include <math.h>

typedef __attribute__((ext_vector_type(16))) __bf16 v16bf;
typedef __attribute__((ext_vector_type(8)))  __bf16 v8bf;
typedef __attribute__((ext_vector_type(8)))  float  v8f;
typedef __attribute__((ext_vector_type(4)))  float  v4f;

#define NEG_INF_F (-1e30f)

constexpr int Bn = 64;    // batch
constexpr int D  = 128;   // feature dim
constexpr int LC = 1024;  // context length
constexpr int LQ = 256;   // query length

// ---------------------------------------------------------------------------
// Fragment loaders from K-contiguous bf16 rows (all 128-bit vector loads).
// A-fragment (16-bit 16x32):             half=0 -> K 0..15 ; half=1 -> K 16..31
// B-fragment (16-bit 32x16):             half=0 -> K 0..15 ; half=1 -> K 16..31
// ---------------------------------------------------------------------------
__device__ __forceinline__ v16bf loadA(const __bf16* row, int k0, int half) {
  return *(const v16bf*)(row + k0 + 16 * half);
}
__device__ __forceinline__ v16bf loadB(const __bf16* row, int k0, int half) {
  return *(const v16bf*)(row + k0 + 16 * half);
}

// ---------------------------------------------------------------------------
// Kernel: cw1[b,i] = sum_k C[b,k,i]*w1[k];  qw2[b,j] = sum_k Q[b,k,j]*w2[k]
// ---------------------------------------------------------------------------
__global__ void k_proj(const float* __restrict__ C, const float* __restrict__ Q,
                       const float* __restrict__ w,
                       float* __restrict__ cw1, float* __restrict__ qw2) {
  int idx = blockIdx.x * blockDim.x + threadIdx.x;
  if (idx < Bn * LC) {
    int b = idx >> 10, i = idx & (LC - 1);
    const float* Cb = C + (size_t)b * D * LC + i;
    float s = 0.f;
#pragma unroll 8
    for (int k = 0; k < D; ++k) s += Cb[(size_t)k * LC] * w[k];
    cw1[idx] = s;
  } else if (idx < Bn * LC + Bn * LQ) {
    int t = idx - Bn * LC;
    int b = t >> 8, j = t & (LQ - 1);
    const float* Qb = Q + (size_t)b * D * LQ + j;
    const float* w2 = w + D;
    float s = 0.f;
#pragma unroll 8
    for (int k = 0; k < D; ++k) s += Qb[(size_t)k * LQ] * w2[k];
    qw2[t] = s;
  }
}

// ---------------------------------------------------------------------------
// Stage C: Cbf[b][k][i] = bf16(C) (native, K-of-k_T's-B is i-contiguous)
//          Abf[b][i][k] = bf16(C[b,k,i]*w3[k])  (LDS-tiled 32x32 transpose)
// ---------------------------------------------------------------------------
__global__ void k_stage_c(const float* __restrict__ C, const float* __restrict__ w,
                          __bf16* __restrict__ Cbf, __bf16* __restrict__ Abf) {
  __shared__ float tile[32][33];
  int bid = blockIdx.x;
  int itile = bid & 31; bid >>= 5;   // LC/32
  int ktile = bid & 3;  bid >>= 2;   // D/32
  int b = bid;
  int tx = threadIdx.x & 31;
  int ty = threadIdx.x >> 5;         // 0..7
  int i0 = itile * 32, k0 = ktile * 32;
  const float* Cb = C + (size_t)b * D * LC;
  __bf16* Cbfb = Cbf + (size_t)b * D * LC;
  __bf16* Abfb = Abf + (size_t)b * LC * D;
  const float* w3 = w + 2 * D;
#pragma unroll
  for (int r = 0; r < 4; ++r) {
    int k = k0 + ty + 8 * r;
    float v = Cb[(size_t)k * LC + i0 + tx];
    tile[ty + 8 * r][tx] = v;
    Cbfb[(size_t)k * LC + i0 + tx] = (__bf16)v;
  }
  __syncthreads();
#pragma unroll
  for (int r = 0; r < 4; ++r) {
    int i = i0 + ty + 8 * r;
    int k = k0 + tx;
    Abfb[(size_t)i * D + k] = (__bf16)(tile[tx][ty + 8 * r] * w3[k]);
  }
}

// ---------------------------------------------------------------------------
// Stage Q: Qbf[b][k][j] = bf16(Q) (native);  Qtbf[b][j][k] = bf16(Q^T)
// ---------------------------------------------------------------------------
__global__ void k_stage_q(const float* __restrict__ Q,
                          __bf16* __restrict__ Qbf, __bf16* __restrict__ Qtbf) {
  __shared__ float tile[32][33];
  int bid = blockIdx.x;
  int jtile = bid & 7; bid >>= 3;    // LQ/32
  int ktile = bid & 3; bid >>= 2;    // D/32
  int b = bid;
  int tx = threadIdx.x & 31;
  int ty = threadIdx.x >> 5;
  int j0 = jtile * 32, k0 = ktile * 32;
  const float* Qb = Q + (size_t)b * D * LQ;
  __bf16* Qbfb = Qbf + (size_t)b * D * LQ;
  __bf16* Qtbfb = Qtbf + (size_t)b * LQ * D;
#pragma unroll
  for (int r = 0; r < 4; ++r) {
    int k = k0 + ty + 8 * r;
    float v = Qb[(size_t)k * LQ + j0 + tx];
    tile[ty + 8 * r][tx] = v;
    Qbfb[(size_t)k * LQ + j0 + tx] = (__bf16)v;
  }
  __syncthreads();
#pragma unroll
  for (int r = 0; r < 4; ++r) {
    int j = j0 + ty + 8 * r;
    Qtbfb[(size_t)j * D + k0 + tx] = (__bf16)tile[tx][ty + 8 * r];
  }
}

// ---------------------------------------------------------------------------
// S[b,i,j] = cw1 + qw2 + Abf @ Qtbf^T  (bf16 WMMA, f32 acc)
// block = 128 = 4 waves; wave -> 16(i) x 16(j) tile; all fragment loads b128
// ---------------------------------------------------------------------------
__global__ void k_score(const __bf16* __restrict__ Abf, const __bf16* __restrict__ Qtbf,
                        const float* __restrict__ cw1, const float* __restrict__ qw2,
                        float* __restrict__ S) {
  int bid = blockIdx.x;
  int jt = bid & 3;  bid >>= 2;          // LQ/64 = 4
  int it = bid & 63; bid >>= 6;          // LC/16 = 64
  int b  = bid;
  int wave = threadIdx.x >> 5;
  int lane = threadIdx.x & 31;
  int m = lane & 15, half = lane >> 4;
  int i0 = it * 16;
  int j0 = jt * 64 + wave * 16;
  const __bf16* Arow = Abf + ((size_t)b * LC + (i0 + m)) * D;
  const __bf16* Brow = Qtbf + ((size_t)b * LQ + (j0 + m)) * D;

  v8f acc = {};
#pragma unroll
  for (int kc = 0; kc < D / 32; ++kc) {
    v16bf a = loadA(Arow, kc * 32, half);
    v16bf bf = loadB(Brow, kc * 32, half);
    acc = __builtin_amdgcn_wmma_f32_16x16x32_bf16(false, a, false, bf,
                                                  (short)0, acc, false, false);
  }
  float q2 = qw2[b * LQ + j0 + m];
#pragma unroll
  for (int r = 0; r < 8; ++r) {
    int i = i0 + r + half * 8;
    S[((size_t)b * LC + i) * LQ + (j0 + m)] = acc[r] + cw1[b * LC + i] + q2;
  }
}

// ---------------------------------------------------------------------------
// Row (over j) masked softmax stats; one wave per (b,i); b128 loads
// ---------------------------------------------------------------------------
__global__ void k_rowstats(const float* __restrict__ S, const float* __restrict__ qmask,
                           float* __restrict__ rmax, float* __restrict__ rinv) {
  int row = blockIdx.x * 8 + (threadIdx.x >> 5);
  int lane = threadIdx.x & 31;
  int b = row >> 10;
  const float* Srow = S + (size_t)row * LQ + lane * 8;
  const float* qm = qmask + b * LQ + lane * 8;
  v4f s0 = *(const v4f*)(Srow);
  v4f s1 = *(const v4f*)(Srow + 4);
  v4f q0 = *(const v4f*)(qm);
  v4f q1 = *(const v4f*)(qm + 4);
  float v[8];
  float mx = NEG_INF_F;
#pragma unroll
  for (int t = 0; t < 4; ++t) {
    v[t]     = s0[t] + (1.0f - q0[t]) * NEG_INF_F;
    v[4 + t] = s1[t] + (1.0f - q1[t]) * NEG_INF_F;
  }
#pragma unroll
  for (int t = 0; t < 8; ++t) mx = fmaxf(mx, v[t]);
#pragma unroll
  for (int off = 16; off > 0; off >>= 1) mx = fmaxf(mx, __shfl_xor(mx, off, 32));
  float s = 0.f;
#pragma unroll
  for (int t = 0; t < 8; ++t) s += __expf(v[t] - mx);
#pragma unroll
  for (int off = 16; off > 0; off >>= 1) s += __shfl_xor(s, off, 32);
  if (lane == 0) { rmax[row] = mx; rinv[row] = 1.0f / s; }
}

// ---------------------------------------------------------------------------
// Column (over i) masked softmax stats, coalesced online logsumexp.
// block 256: lanes along j (coalesced), 8 i-strips, LDS merge.
// ---------------------------------------------------------------------------
__global__ void k_colstats(const float* __restrict__ S, const float* __restrict__ cmask,
                           float* __restrict__ cmax, float* __restrict__ cinv) {
  __shared__ float smax[8][32];
  __shared__ float ssum[8][32];
  int bid = blockIdx.x;
  int jt = bid & 7; int b = bid >> 3;    // LQ/32 = 8
  int tx = threadIdx.x & 31, ty = threadIdx.x >> 5;
  int j = jt * 32 + tx;
  const float* Sb = S + (size_t)b * LC * LQ;
  const float* cm = cmask + b * LC;
  float m = NEG_INF_F, s = 0.f;
  for (int i = ty; i < LC; i += 8) {
    float v = Sb[(size_t)i * LQ + j] + (1.0f - cm[i]) * NEG_INF_F;
    float mn = fmaxf(m, v);
    s = s * __expf(m - mn) + __expf(v - mn);
    m = mn;
  }
  smax[ty][tx] = m; ssum[ty][tx] = s;
  __syncthreads();
  if (ty == 0) {
    float M = smax[0][tx], Sm = ssum[0][tx];
#pragma unroll
    for (int t = 1; t < 8; ++t) {
      float m2 = smax[t][tx], s2 = ssum[t][tx];
      float mn = fmaxf(M, m2);
      Sm = Sm * __expf(M - mn) + s2 * __expf(m2 - mn);
      M = mn;
    }
    cmax[b * LQ + j] = M;
    cinv[b * LQ + j] = 1.0f / Sm;
  }
}

// ---------------------------------------------------------------------------
// S1bf[b][i][j] = bf16(exp(S+mask-rmax)*rinv)  (elementwise, b128 in/out)
// ---------------------------------------------------------------------------
__global__ void k_s1bf(const float* __restrict__ S, const float* __restrict__ qmask,
                       const float* __restrict__ rmax, const float* __restrict__ rinv,
                       __bf16* __restrict__ S1bf) {
  int t = blockIdx.x * blockDim.x + threadIdx.x;
  int lane = t & 31;                 // j-octet
  int row = t >> 5;                  // (b,i)
  int b = row >> 10;
  int j0 = lane * 8;
  const float* Srow = S + (size_t)row * LQ + j0;
  const float* qm = qmask + b * LQ + j0;
  float mx = rmax[row], iv = rinv[row];
  v4f a = *(const v4f*)(Srow);
  v4f c = *(const v4f*)(Srow + 4);
  v4f q0 = *(const v4f*)(qm);
  v4f q1 = *(const v4f*)(qm + 4);
  v8bf o;
#pragma unroll
  for (int e = 0; e < 4; ++e) {
    o[e]     = (__bf16)(__expf(a[e] + (1.0f - q0[e]) * NEG_INF_F - mx) * iv);
    o[4 + e] = (__bf16)(__expf(c[e] + (1.0f - q1[e]) * NEG_INF_F - mx) * iv);
  }
  *(v8bf*)(S1bf + (size_t)row * LQ + j0) = o;
}

// ---------------------------------------------------------------------------
// S2bf[b][j][i] = bf16(exp(S+mask-cmax)*cinv) transposed (LDS 32x32 tiles)
// ---------------------------------------------------------------------------
__global__ void k_s2t(const float* __restrict__ S, const float* __restrict__ cmask,
                      const float* __restrict__ cmax, const float* __restrict__ cinv,
                      __bf16* __restrict__ S2bf) {
  __shared__ float tile[32][33];
  int bid = blockIdx.x;
  int jt = bid & 7;  bid >>= 3;      // LQ/32
  int it = bid & 31; bid >>= 5;      // LC/32
  int b = bid;
  int tx = threadIdx.x & 31, ty = threadIdx.x >> 5;
  int i0 = it * 32, j0 = jt * 32;
  const float* Sb = S + (size_t)b * LC * LQ;
  const float* cm = cmask + b * LC;
  float cmx = cmax[b * LQ + j0 + tx];
  float civ = cinv[b * LQ + j0 + tx];
#pragma unroll
  for (int r = 0; r < 4; ++r) {
    int i = i0 + ty + 8 * r;
    float v = Sb[(size_t)i * LQ + j0 + tx] + (1.0f - cm[i]) * NEG_INF_F;
    tile[ty + 8 * r][tx] = __expf(v - cmx) * civ;
  }
  __syncthreads();
  __bf16* out = S2bf + (size_t)b * LQ * LC;
#pragma unroll
  for (int r = 0; r < 4; ++r) {
    int j = j0 + ty + 8 * r;
    out[(size_t)j * LC + i0 + tx] = (__bf16)tile[tx][ty + 8 * r];
  }
}

// ---------------------------------------------------------------------------
// Tbf[b][dd][j] = (S2^T @ Ct)^T  via WMMA (K = Lc = 1024), b128 operand loads,
// D-fragment rows are contiguous j -> packed v8bf store (transposed layout).
// ---------------------------------------------------------------------------
__global__ void k_T(const __bf16* __restrict__ S2bf, const __bf16* __restrict__ Cbf,
                    __bf16* __restrict__ Tbf) {
  int bid = blockIdx.x;
  int tg = bid & 31; int b = bid >> 5;
  int wave = threadIdx.x >> 5;
  int tile = tg * 4 + wave;                 // 0..127
  int jt = tile >> 3, ddt = tile & 7;
  int lane = threadIdx.x & 31;
  int m = lane & 15, half = lane >> 4;
  int j0 = jt * 16, dd0 = ddt * 16;
  const __bf16* Arow = S2bf + ((size_t)b * LQ + (j0 + m)) * LC;   // A[j][i]
  const __bf16* Brow = Cbf + ((size_t)b * D + (dd0 + m)) * LC;    // B[i][dd]
  v8f acc = {};
  for (int kc = 0; kc < LC / 32; ++kc) {
    v16bf a = loadA(Arow, kc * 32, half);
    v16bf bf = loadB(Brow, kc * 32, half);
    acc = __builtin_amdgcn_wmma_f32_16x16x32_bf16(false, a, false, bf,
                                                  (short)0, acc, false, false);
  }
  v8bf o;
#pragma unroll
  for (int r = 0; r < 8; ++r) o[r] = (__bf16)acc[r];
  *(v8bf*)(Tbf + ((size_t)b * D + dd0 + m) * LQ + j0 + 8 * half) = o;
}

// ---------------------------------------------------------------------------
// A = S1@Qt, Bm = S1@T; out = [Ct | A | Ct*A | Ct*Bm] as (B,4d,Lc).
// All operand loads b128; epilogue C loads + all stores are v4f (b128).
// ---------------------------------------------------------------------------
__global__ void k_out(const __bf16* __restrict__ S1bf, const float* __restrict__ C,
                      const __bf16* __restrict__ Qbf, const __bf16* __restrict__ Tbf,
                      float* __restrict__ out) {
  int bid = blockIdx.x;
  int it = bid & 63; int b = bid >> 6;
  int wave = threadIdx.x >> 5;
  int lane = threadIdx.x & 31;
  int m = lane & 15, half = lane >> 4;
  int i0 = it * 16;
  const __bf16* Arow = S1bf + ((size_t)b * LC + (i0 + m)) * LQ;   // A[i][j]
  const __bf16* Qb = Qbf + (size_t)b * D * LQ;
  const __bf16* Tb = Tbf + (size_t)b * D * LQ;
  const float* Cb = C + (size_t)b * D * LC;
  float* ob = out + (size_t)b * 4 * D * LC;

  v8f accA[2] = {{}, {}};
  v8f accB[2] = {{}, {}};
#pragma unroll
  for (int kc = 0; kc < LQ / 32; ++kc) {
    v16bf s1 = loadA(Arow, kc * 32, half);
#pragma unroll
    for (int tt = 0; tt < 2; ++tt) {
      int dd = (wave * 2 + tt) * 16 + m;
      v16bf bq = loadB(Qb + (size_t)dd * LQ, kc * 32, half);      // B[j][dd]=Qt[j,dd]
      v16bf bt = loadB(Tb + (size_t)dd * LQ, kc * 32, half);      // B[j][dd]=T[j,dd]
      accA[tt] = __builtin_amdgcn_wmma_f32_16x16x32_bf16(false, s1, false, bq,
                                                         (short)0, accA[tt], false, false);
      accB[tt] = __builtin_amdgcn_wmma_f32_16x16x32_bf16(false, s1, false, bt,
                                                         (short)0, accB[tt], false, false);
    }
  }
#pragma unroll
  for (int tt = 0; tt < 2; ++tt) {
    int dd = (wave * 2 + tt) * 16 + m;
    int ib = i0 + 8 * half;                       // 8 consecutive i per lane
    const float* cp = Cb + (size_t)dd * LC + ib;
    v4f c0 = *(const v4f*)(cp);
    v4f c1 = *(const v4f*)(cp + 4);
    v4f a0, a1, b0, b1, ca0, ca1, cb0, cb1;
#pragma unroll
    for (int r = 0; r < 4; ++r) {
      a0[r] = accA[tt][r];      a1[r] = accA[tt][4 + r];
      b0[r] = accB[tt][r];      b1[r] = accB[tt][4 + r];
      ca0[r] = c0[r] * a0[r];   ca1[r] = c1[r] * a1[r];
      cb0[r] = c0[r] * b0[r];   cb1[r] = c1[r] * b1[r];
    }
    float* p0 = ob + (size_t)dd * LC + ib;
    float* p1 = ob + (size_t)(D + dd) * LC + ib;
    float* p2 = ob + (size_t)(2 * D + dd) * LC + ib;
    float* p3 = ob + (size_t)(3 * D + dd) * LC + ib;
    *(v4f*)(p0) = c0;  *(v4f*)(p0 + 4) = c1;
    *(v4f*)(p1) = a0;  *(v4f*)(p1 + 4) = a1;
    *(v4f*)(p2) = ca0; *(v4f*)(p2 + 4) = ca1;
    *(v4f*)(p3) = cb0; *(v4f*)(p3 + 4) = cb1;
  }
}

// ---------------------------------------------------------------------------
extern "C" void kernel_launch(void* const* d_in, const int* in_sizes, int n_in,
                              void* d_out, int out_size, void* d_ws, size_t ws_size,
                              hipStream_t stream) {
  const float* C     = (const float*)d_in[0];
  const float* Q     = (const float*)d_in[1];
  const float* cmask = (const float*)d_in[2];
  const float* qmask = (const float*)d_in[3];
  const float* w     = (const float*)d_in[4];
  float* out = (float*)d_out;

  char* wsb = (char*)d_ws;
  size_t off = 0;
  auto alloc = [&](size_t bytes) -> void* {
    off = (off + 255) & ~(size_t)255;
    void* p = wsb + off;
    off += bytes;
    return p;
  };
  float*  S     = (float*) alloc((size_t)Bn * LC * LQ * 4);   // 67.1 MB
  __bf16* S1bf  = (__bf16*)alloc((size_t)Bn * LC * LQ * 2);   // 33.5 MB
  __bf16* S2bf  = (__bf16*)alloc((size_t)Bn * LQ * LC * 2);   // 33.5 MB
  __bf16* Abf   = (__bf16*)alloc((size_t)Bn * LC * D * 2);    // 16.8 MB
  __bf16* Cbf   = (__bf16*)alloc((size_t)Bn * D * LC * 2);    // 16.8 MB
  __bf16* Qtbf  = (__bf16*)alloc((size_t)Bn * LQ * D * 2);    //  4.2 MB
  __bf16* Qbf   = (__bf16*)alloc((size_t)Bn * D * LQ * 2);    //  4.2 MB
  __bf16* Tbf   = (__bf16*)alloc((size_t)Bn * D * LQ * 2);    //  4.2 MB
  float*  cw1   = (float*) alloc((size_t)Bn * LC * 4);
  float*  qw2   = (float*) alloc((size_t)Bn * LQ * 4);
  float*  rmax  = (float*) alloc((size_t)Bn * LC * 4);
  float*  rinv  = (float*) alloc((size_t)Bn * LC * 4);
  float*  cmax  = (float*) alloc((size_t)Bn * LQ * 4);
  float*  cinv  = (float*) alloc((size_t)Bn * LQ * 4);

  k_proj<<<(Bn * LC + Bn * LQ + 255) / 256, 256, 0, stream>>>(C, Q, w, cw1, qw2);
  k_stage_c<<<Bn * (D / 32) * (LC / 32), 256, 0, stream>>>(C, w, Cbf, Abf);
  k_stage_q<<<Bn * (D / 32) * (LQ / 32), 256, 0, stream>>>(Q, Qbf, Qtbf);
  k_score<<<Bn * (LC / 16) * (LQ / 64), 128, 0, stream>>>(Abf, Qtbf, cw1, qw2, S);
  k_rowstats<<<(Bn * LC) / 8, 256, 0, stream>>>(S, qmask, rmax, rinv);
  k_colstats<<<Bn * (LQ / 32), 256, 0, stream>>>(S, cmask, cmax, cinv);
  k_s1bf<<<(Bn * LC * (LQ / 8)) / 256, 256, 0, stream>>>(S, qmask, rmax, rinv, S1bf);
  k_s2t<<<Bn * (LC / 32) * (LQ / 32), 256, 0, stream>>>(S, cmask, cmax, cinv, S2bf);
  k_T<<<Bn * 32, 128, 0, stream>>>(S2bf, Cbf, Tbf);
  k_out<<<Bn * (LC / 16), 128, 0, stream>>>(S1bf, C, Qbf, Tbf, out);
}
